// HyperComputeModule_32813550141848
// MI455X (gfx1250) — matrix operand order repaired and checked by reading, not compiled, not run on hardware
//
#include <hip/hip_runtime.h>
#include <stdint.h>

typedef __attribute__((ext_vector_type(16))) _Float16 v16h;
typedef __attribute__((ext_vector_type(8)))  float    v8f;
typedef __attribute__((ext_vector_type(4)))  unsigned int u32x4;
typedef __attribute__((ext_vector_type(8)))  int      i32x8;
typedef __attribute__((ext_vector_type(4)))  int      i32x4;

#define B_ 8
#define C_ 256
#define D_ 2048
#define THRESH2 484.0f   // 22^2 ; dist<22  <=>  d2<484 (sqrt monotone, 484>0)
#define LN_EPS_ 1e-5f

#if __has_builtin(__builtin_amdgcn_tensor_load_to_lds)
#define USE_TDM 1
#else
#define USE_TDM 0
#endif

#define WMMA_F16(A,Bv,Cv) __builtin_amdgcn_wmma_f32_16x16x32_f16(false,(A),false,(Bv),(short)0,(Cv),false,false)

static __device__ __forceinline__ int laneid() { return threadIdx.x & 31; }

// ---- CDNA5 async global->LDS copy (ASYNCcnt path), 16 bytes per lane per issue ----
static __device__ __forceinline__ void async_b128(uint32_t lds_addr, const void* gaddr) {
  asm volatile("global_load_async_to_lds_b128 %0, %1, off"
               :: "v"(lds_addr), "v"(gaddr) : "memory");
}
static __device__ __forceinline__ void wait_async0() {
  asm volatile("s_wait_asynccnt 0x0" ::: "memory");
}

#if USE_TDM
// ---- CDNA5 Tensor Data Mover: 1-D contiguous tile (n8 x 8B) global -> LDS, TENSORcnt path ----
// D# per CDNA5 ISA: group0 = {count=1 | lds_addr | global_addr | type=2},
// group1 = {data_size=8B, tensor_dim0, tensor_dim1=1, tile_dim0=n8, contiguous stride}.
// This toolchain carries the 6-arg builtin: (u32x4, i32x8, i32x4, i32x4, i32x8, i32 cpol).
static __device__ __forceinline__ void tdm_load_1d(uint32_t lds_addr, const void* gaddr,
                                                   unsigned n8, unsigned tensor_n8) {
  uint64_t ga = (uint64_t)(uintptr_t)gaddr;
  u32x4 g0;
  g0[0] = 1u;                                                   // count=1, user descriptor
  g0[1] = lds_addr;                                             // D#.lds_addr
  g0[2] = (uint32_t)ga;                                         // global_addr[31:0]
  g0[3] = (uint32_t)((ga >> 32) & 0x01FFFFFFu) | (2u << 30);    // global_addr[56:32] | type=2
  i32x8 g1;
  g1[0] = (int)(3u << 16);                                      // data_size=8B; no mcast/pad/iter
  g1[1] = (int)((tensor_n8 & 0xFFFFu) << 16);                   // tensor_dim0[15:0]
  g1[2] = (int)((tensor_n8 >> 16) | (1u << 16));                // tensor_dim0[31:16] | tensor_dim1=1
  g1[3] = (int)((n8 & 0xFFFFu) << 16);                          // tile_dim0 = n8 (1-D tile)
  g1[4] = 0;                                                    // tile_dim1/2 unused
  g1[5] = (int)tensor_n8;                                       // tensor_dim0_stride (contiguous)
  g1[6] = 0;
  g1[7] = 0;
  i32x4 z4 = {0, 0, 0, 0};                                      // groups 2/3 unused (<=2-D tensor)
  i32x8 z8 = {0, 0, 0, 0, 0, 0, 0, 0};
  __builtin_amdgcn_tensor_load_to_lds(g0, g1, z4, z4, z8, 0);
}
static __device__ __forceinline__ void wait_tensor0() {
  __builtin_amdgcn_s_wait_tensorcnt((short)0);
}
#endif

// Fragment-ready swizzled layout for WMMA B operands over a [K x 256] f16 matrix:
// element (k,n) -> ((k>>5)*16 + (n>>4))*512 + ((n&15) + (k&16))*16 + (k&15)
// => for a 32-k block and 16-wide n-tile, lane l's v16h fragment is 32 contiguous bytes:
//    fragment[h] = element(k = k0 + (l&16) + h, n = n0 + (l&15))   -- contiguous k order, same as the A fragments.
static __device__ __forceinline__ size_t bswz(int k, int n) {
  return ((size_t)((k >> 5) * 16 + (n >> 4)) << 9) + (size_t)(((n & 15) + (k & 16)) << 4) + (k & 15);
}

// A fragment (16x32 f16) from row-major [M x K], tile at (m0,k0)
static __device__ __forceinline__ v16h load_a_rm(const _Float16* base, int lda, int m0, int k0) {
  int lane = laneid();
  const _Float16* row = base + (size_t)(m0 + (lane & 15)) * lda + k0 + ((lane >> 4) << 4);
  v16h a;
#pragma unroll
  for (int h = 0; h < 16; ++h) a[h] = row[h];
  return a;
}

// B fragment (32x16 f16) with B[k][n] = src[n][k]  (src row-major [N x K])
static __device__ __forceinline__ v16h load_b_tr(const _Float16* src, int ld, int k0, int n0) {
  int lane = laneid();
  const _Float16* row = src + (size_t)(n0 + (lane & 15)) * ld + k0 + ((lane >> 4) << 4);
  v16h b;
#pragma unroll
  for (int h = 0; h < 16; ++h) b[h] = row[h];
  return b;
}

// B fragment from a swizzled LDS panel (one 32-k block = 8192 halves): 32 contiguous bytes/lane
static __device__ __forceinline__ v16h load_b_lds_swz(const _Float16* panel, int n0) {
  const _Float16* p = panel + ((n0 >> 4) << 9) + (laneid() << 4);
  return *(const v16h*)p;     // -> ds_load_b128 x2
}

// expand 32 incidence bits into an exact {0,1} f16 A fragment
static __device__ __forceinline__ v16h expand_bits(unsigned w, int kofs) {
  v16h a;
#pragma unroll
  for (int h = 0; h < 16; ++h) {
    int bit = h + kofs;
    a[h] = (_Float16)(float)((w >> bit) & 1u);
  }
  return a;
}

// ---------------- K1: transpose x[b,c,d] -> xt[b,d,c] (f32) + hi/lo f16 split ----------------
__global__ void k_transpose_split(const float* __restrict__ x, float* __restrict__ xt,
                                  _Float16* __restrict__ xhi, _Float16* __restrict__ xlo) {
  __shared__ float tile[32][33];
  int b = blockIdx.z;
  int d0 = blockIdx.x * 32, c0 = blockIdx.y * 32;
  int tx = threadIdx.x, ty = threadIdx.y;
#pragma unroll
  for (int j = 0; j < 32; j += 8)
    tile[ty + j][tx] = x[((size_t)b * C_ + (c0 + ty + j)) * D_ + (d0 + tx)];
  __syncthreads();
#pragma unroll
  for (int j = 0; j < 32; j += 8) {
    int d = d0 + ty + j, c = c0 + tx;
    float v = tile[tx][ty + j];
    size_t o = ((size_t)b * D_ + d) * C_ + c;
    xt[o] = v;
    _Float16 hi = (_Float16)v;
    xhi[o] = hi;
    xlo[o] = (_Float16)(v - (float)hi);
  }
}

// ---------------- K1b: sq[row] = sum_c xt[row][c]^2 (one wave per row) ----------------
__global__ void k_rowsq(const float* __restrict__ xt, float* __restrict__ sq) {
  int row = blockIdx.x * 8 + (threadIdx.x >> 5);
  int lane = threadIdx.x & 31;
  const float* r = xt + (size_t)row * C_;
  float s = 0.f;
#pragma unroll
  for (int i = 0; i < 8; ++i) { float v = r[lane + 32 * i]; s += v * v; }
#pragma unroll
  for (int off = 16; off; off >>= 1) s += __shfl_down(s, off, 32);
  if (lane == 0) sq[row] = s;
}

// ---------------- K2a: split W_fc into hi/lo f16 ----------------
__global__ void k_splitW(const float* __restrict__ W, _Float16* __restrict__ whi,
                         _Float16* __restrict__ wlo) {
  int i = blockIdx.x * 256 + threadIdx.x;
  float v = W[i];
  _Float16 hi = (_Float16)v;
  whi[i] = hi;
  wlo[i] = (_Float16)(v - (float)hi);
}

// ---------------- K2: xf = xt @ W^T + b (M=16384,N=256,K=256); output in swizzled layout ------
__global__ void k_fc(const _Float16* __restrict__ xhi, const _Float16* __restrict__ xlo,
                     const _Float16* __restrict__ whi, const _Float16* __restrict__ wlo,
                     const float* __restrict__ bias,
                     _Float16* __restrict__ fhi, _Float16* __restrict__ flo) {
  int wave = threadIdx.x >> 5, lane = laneid();
  int m0 = blockIdx.x * 16;
  int n0 = (blockIdx.y * 8 + wave) * 16;
  v8f acc = {};
#pragma unroll
  for (int k0 = 0; k0 < C_; k0 += 32) {
    v16h ah = load_a_rm(xhi, C_, m0, k0);
    v16h al = load_a_rm(xlo, C_, m0, k0);
    v16h bh = load_b_tr(whi, C_, k0, n0);   // B[k][n] = W[n][k]
    v16h bl = load_b_tr(wlo, C_, k0, n0);
    acc = WMMA_F16(ah, bh, acc);
    acc = WMMA_F16(ah, bl, acc);
    acc = WMMA_F16(al, bh, acc);
  }
  float bn = bias[n0 + (lane & 15)];
  int mofs = (lane >> 4) << 3;
#pragma unroll
  for (int e = 0; e < 8; ++e) {
    int row = m0 + e + mofs;                // flat over B*D
    int bb = row >> 11, d = row & (D_ - 1);
    int n = n0 + (lane & 15);
    size_t o = (size_t)bb * D_ * C_ + bswz(d, n);
    float v = acc[e] + bn;
    _Float16 hi = (_Float16)v;
    fhi[o] = hi;
    flo[o] = (_Float16)(v - (float)hi);
  }
}

// ---------------- K3: Gram tile + threshold -> incidence bitmask via ballot ----------------
__global__ void k_gram(const _Float16* __restrict__ xhi, const _Float16* __restrict__ xlo,
                       const float* __restrict__ sq, uint16_t* __restrict__ hg16) {
  int b = blockIdx.z, wave = threadIdx.x >> 5, lane = laneid();
  int i0 = blockIdx.x * 16;
  int j0 = (blockIdx.y * 8 + wave) * 16;
  const _Float16* Ah = xhi + (size_t)b * D_ * C_;
  const _Float16* Al = xlo + (size_t)b * D_ * C_;
  v8f acc = {};
#pragma unroll
  for (int k0 = 0; k0 < C_; k0 += 32) {
    v16h ah = load_a_rm(Ah, C_, i0, k0);
    v16h al = load_a_rm(Al, C_, i0, k0);
    v16h bh = load_b_tr(Ah, C_, k0, j0);    // B[k][n] = xt[j0+n][k]
    v16h bl = load_b_tr(Al, C_, k0, j0);
    acc = WMMA_F16(ah, bh, acc);
    acc = WMMA_F16(ah, bl, acc);
    acc = WMMA_F16(al, bh, acc);
  }
  const float* sqb = sq + (size_t)b * D_;
  float sqj = sqb[j0 + (lane & 15)];
  int mofs = (lane >> 4) << 3;
  int wcol = j0 >> 4;
#pragma unroll
  for (int e = 0; e < 8; ++e) {
    int row = i0 + e + mofs;                 // lanes<16: i0+e ; lanes>=16: i0+8+e
    float d2 = sqb[row] + sqj - 2.0f * acc[e];
    unsigned m = (unsigned)__ballot(d2 < THRESH2);
    if (lane == 0)  hg16[((size_t)b * D_ + (i0 + e)) * (D_ / 16) + wcol] = (uint16_t)(m & 0xFFFFu);
    if (lane == 16) hg16[((size_t)b * D_ + (i0 + 8 + e)) * (D_ / 16) + wcol] = (uint16_t)(m >> 16);
  }
}

// ---------------- K4: inv_deg[row] = 1 / popcount(hg row) ----------------
__global__ void k_deg(const uint32_t* __restrict__ hg32, float* __restrict__ invdeg) {
  int row = blockIdx.x * 8 + (threadIdx.x >> 5);
  int lane = threadIdx.x & 31;
  const uint32_t* r = hg32 + (size_t)row * (D_ / 32);
  int cnt = __popc(r[lane]) + __popc(r[lane + 32]);
#pragma unroll
  for (int off = 16; off; off >>= 1) cnt += __shfl_down(cnt, off, 32);
  if (lane == 0) invdeg[row] = 1.0f / (float)cnt;
}

// ---------------- K5/K6: mean aggregation GEMM, TDM + async-LDS double-buffered panels -------
// MODE 0: out = (hg @ (Bh+Bl)) * invdeg       -> swizzled outHi/outLo (f16)
// MODE 1: y   = (hg @ (Bh+Bl)) * invdeg + xt  -> f32 [b,d,c]
// Block: 8 waves, 32-row m-tile x N=256 (wave w owns n-tiles w*32, w*32+16; 8 WMMA/step).
// Per 32-k step: hi panel (16KB) staged by ONE tensor_load_to_lds (TDM, wave 0, TENSORcnt);
// lo panel (16KB) staged by global_load_async_to_lds_b128 (all waves, ASYNCcnt); double-buffered.
template <int MODE>
__global__ void k_agg(const uint32_t* __restrict__ hg32,
                      const _Float16* __restrict__ Bhg, const _Float16* __restrict__ Blg,
                      const float* __restrict__ invdeg, const float* __restrict__ xt,
                      _Float16* __restrict__ outHi, _Float16* __restrict__ outLo,
                      float* __restrict__ yout) {
  __shared__ __align__(32) _Float16 panel[2][2][32 * C_];   // [buf][hi/lo][8192] = 64 KB
  int b = blockIdx.z, wave = threadIdx.x >> 5, lane = laneid();
  int t = threadIdx.x;
  int m0 = blockIdx.x * 32;
  int n0 = wave * 32;

  const uint32_t* rowbits0 = hg32 + ((size_t)b * D_ + m0 + (lane & 15)) * (D_ / 32);
  const uint32_t* rowbits1 = rowbits0 + 16 * (D_ / 32);
  const char* gHi = (const char*)(Bhg + (size_t)b * D_ * C_);
  const char* gLo = (const char*)(Blg + (size_t)b * D_ * C_);
  uint32_t ldsHi[2], ldsLo[2];
#pragma unroll
  for (int u = 0; u < 2; ++u) {
    ldsHi[u] = (uint32_t)(uintptr_t)&panel[u][0][0];
    ldsLo[u] = (uint32_t)(uintptr_t)&panel[u][1][0];
  }

  auto stage = [&](int step, int buf) {
    size_t pbase = (size_t)step * 32 * C_ * 2;   // swizzled panel is contiguous 16KB (x2)
#if USE_TDM
    if (wave == 0)                               // one TDM descriptor moves the whole hi panel
      tdm_load_1d(ldsHi[buf], gHi + pbase, /*tile 8B units*/ (32 * C_ * 2) / 8,
                  /*tensor 8B units*/ (unsigned)(((size_t)D_ * C_ * 2) / 8));
#else
#pragma unroll
    for (int i = 0; i < 4; ++i) {
      uint32_t boff = (uint32_t)(i * 256 + t) * 16;
      async_b128(ldsHi[buf] + boff, gHi + pbase + boff);
    }
#endif
#pragma unroll
    for (int i = 0; i < 4; ++i) {
      uint32_t boff = (uint32_t)(i * 256 + t) * 16;
      async_b128(ldsLo[buf] + boff, gLo + pbase + boff);
    }
  };

  int kofs = (lane >> 4) << 4;
  v8f acc00 = {}, acc01 = {}, acc10 = {}, acc11 = {};
  stage(0, 0);
  int cur = 0;
  for (int step = 0; step < D_ / 32; ++step) {
    wait_async0();                    // own wave's async (lo-panel) loads complete
#if USE_TDM
    if (wave == 0) wait_tensor0();    // hi-panel TDM complete (wave 0 issued it)
#endif
    __syncthreads();                  // whole block sees both panels resident
    if (step + 1 < D_ / 32) stage(step + 1, cur ^ 1);
    __builtin_prefetch((const void*)(rowbits0 + step + 8), 0, 0);   // global_prefetch_b8

    v16h a0 = expand_bits(rowbits0[step], kofs);
    v16h a1 = expand_bits(rowbits1[step], kofs);
    const _Float16* pHi = &panel[cur][0][0];
    const _Float16* pLo = &panel[cur][1][0];
    v16h bh0 = load_b_lds_swz(pHi, n0);
    v16h bl0 = load_b_lds_swz(pLo, n0);
    v16h bh1 = load_b_lds_swz(pHi, n0 + 16);
    v16h bl1 = load_b_lds_swz(pLo, n0 + 16);
    acc00 = WMMA_F16(a0, bh0, acc00);
    acc00 = WMMA_F16(a0, bl0, acc00);
    acc01 = WMMA_F16(a0, bh1, acc01);
    acc01 = WMMA_F16(a0, bl1, acc01);
    acc10 = WMMA_F16(a1, bh0, acc10);
    acc10 = WMMA_F16(a1, bl0, acc10);
    acc11 = WMMA_F16(a1, bh1, acc11);
    acc11 = WMMA_F16(a1, bl1, acc11);

    __syncthreads();                  // all reads of panel[cur] done before it is restaged
    cur ^= 1;
  }

  const float* idb = invdeg + (size_t)b * D_;
  int mofs = (lane >> 4) << 3;
#pragma unroll
  for (int mt = 0; mt < 2; ++mt) {
#pragma unroll
    for (int e = 0; e < 8; ++e) {
      int row = m0 + mt * 16 + e + mofs;
      float inv = idb[row];
#pragma unroll
      for (int tix = 0; tix < 2; ++tix) {
        float v;
        if (mt == 0) v = (tix ? acc01[e] : acc00[e]) * inv;
        else         v = (tix ? acc11[e] : acc10[e]) * inv;
        int n = n0 + tix * 16 + (lane & 15);
        if (MODE == 0) {
          size_t o = (size_t)b * D_ * C_ + bswz(row, n);
          _Float16 hi = (_Float16)v;
          outHi[o] = hi;
          outLo[o] = (_Float16)(v - (float)hi);
        } else {
          size_t o = ((size_t)b * D_ + row) * C_ + n;
          yout[o] = v + xt[o];
        }
      }
    }
  }
}

// ---------------- K7: LayerNorm over d-axis + SiLU, write out[b,c,d] ----------------
__global__ void k_ln_silu(const float* __restrict__ y, const float* __restrict__ gamma,
                          const float* __restrict__ beta, float* __restrict__ out) {
  int b = blockIdx.x >> 8;        // C_ == 256
  int c = blockIdx.x & 255;
  int t = threadIdx.x;
  float vals[8];
  float s = 0.f, s2 = 0.f;
#pragma unroll
  for (int i = 0; i < 8; ++i) {
    int d = t + 256 * i;
    float v = y[((size_t)b * D_ + d) * C_ + c];
    vals[i] = v; s += v; s2 += v * v;
  }
#pragma unroll
  for (int off = 16; off; off >>= 1) { s += __shfl_down(s, off, 32); s2 += __shfl_down(s2, off, 32); }
  __shared__ float red[2][8];
  __shared__ float stats[2];
  int wave = t >> 5, lane = t & 31;
  if (lane == 0) { red[0][wave] = s; red[1][wave] = s2; }
  __syncthreads();
  if (t == 0) {
    float a = 0.f, q = 0.f;
#pragma unroll
    for (int i = 0; i < 8; ++i) { a += red[0][i]; q += red[1][i]; }
    float mu = a / (float)D_;
    float var = q / (float)D_ - mu * mu;
    stats[0] = mu;
    stats[1] = rsqrtf(var + LN_EPS_);
  }
  __syncthreads();
  float mu = stats[0], rstd = stats[1];
#pragma unroll
  for (int i = 0; i < 8; ++i) {
    int d = t + 256 * i;
    float yn = (vals[i] - mu) * rstd * gamma[d] + beta[d];
    out[((size_t)b * C_ + c) * D_ + d] = yn / (1.0f + __expf(-yn));
  }
}

// ---------------- host launcher ----------------
extern "C" void kernel_launch(void* const* d_in, const int* in_sizes, int n_in,
                              void* d_out, int out_size, void* d_ws, size_t ws_size,
                              hipStream_t stream) {
  const float* x     = (const float*)d_in[0];
  const float* W_fc  = (const float*)d_in[1];
  const float* b_fc  = (const float*)d_in[2];
  const float* gamma = (const float*)d_in[3];
  const float* beta  = (const float*)d_in[4];
  float* out = (float*)d_out;

  const size_t BDC = (size_t)B_ * D_ * C_;   // 4,194,304 elems
  const size_t BD  = (size_t)B_ * D_;        // 16,384
  char* wsb = (char*)d_ws;
  size_t off = 0;
  auto alloc = [&](size_t bytes) -> void* {
    void* p = (void*)(wsb + off);
    off += (bytes + 255) & ~(size_t)255;
    return p;
  };

  float*     xt     = (float*)alloc(BDC * 4);          // 16.8 MB
  _Float16*  xhi    = (_Float16*)alloc(BDC * 2);       //  8.4 MB
  _Float16*  xlo    = (_Float16*)alloc(BDC * 2);       //  8.4 MB
  float*     sq     = (float*)alloc(BD * 4);
  float*     invdeg = (float*)alloc(BD * 4);
  _Float16*  whi    = (_Float16*)alloc((size_t)C_ * C_ * 2);
  _Float16*  wlo    = (_Float16*)alloc((size_t)C_ * C_ * 2);
  _Float16*  fhi    = (_Float16*)alloc(BDC * 2);       //  8.4 MB (swizzled)
  _Float16*  flo    = (_Float16*)alloc(BDC * 2);       //  8.4 MB (swizzled)
  uint16_t*  hg16   = (uint16_t*)alloc((size_t)B_ * D_ * (D_ / 16) * 2);  // 4.2 MB
  _Float16*  ehi    = (_Float16*)alloc(BDC * 2);       //  8.4 MB (swizzled)
  _Float16*  elo    = (_Float16*)alloc(BDC * 2);       //  8.4 MB (swizzled)
  float*     ybuf   = (float*)alloc(BDC * 4);          // 16.8 MB
  const uint32_t* hg32 = (const uint32_t*)hg16;
  (void)ws_size; (void)in_sizes; (void)n_in; (void)out_size;

  k_transpose_split<<<dim3(D_ / 32, C_ / 32, B_), dim3(32, 8), 0, stream>>>(x, xt, xhi, xlo);
  k_rowsq<<<dim3((unsigned)(BD / 8)), 256, 0, stream>>>(xt, sq);
  k_splitW<<<dim3((C_ * C_) / 256), 256, 0, stream>>>(W_fc, whi, wlo);
  k_fc<<<dim3((unsigned)(BD / 16), 2), 256, 0, stream>>>(xhi, xlo, whi, wlo, b_fc, fhi, flo);
  k_gram<<<dim3(D_ / 16, 16, B_), 256, 0, stream>>>(xhi, xlo, sq, hg16);
  k_deg<<<dim3((unsigned)(BD / 8)), 256, 0, stream>>>(hg32, invdeg);
  // E = mean-agg(hg, xf)  (TDM + async-LDS staged swizzled B panels)
  k_agg<0><<<dim3(D_ / 32, 1, B_), 256, 0, stream>>>(hg32, fhi, flo, invdeg, nullptr, ehi, elo, nullptr);
  // y = mean-agg(hg, E) + xt
  k_agg<1><<<dim3(D_ / 32, 1, B_), 256, 0, stream>>>(hg32, ehi, elo, invdeg, xt, nullptr, nullptr, ybuf);
  k_ln_silu<<<dim3((unsigned)(B_ * C_)), 256, 0, stream>>>(ybuf, gamma, beta, out);
}
